// GCU_64476049047913
// MI455X (gfx1250) — compile-verified, matrix operand order checked by reading, not run on hardware
//
#include <hip/hip_runtime.h>
#include <hip/hip_bf16.h>

// ---------------------------------------------------------------------------
// MI455X (gfx1250) GNN message-passing step.
//   curvv = sigmoid(z @ W_cur^T + b_cur)                     [N,64]
//   m     = sigmoid(concat(z[dst],z[src]) @ W_nbr^T + b_nbr) [E,64]
//   nbrvv = segment_sum(m, dst)                              [N,64]
//   out   = tanh(concat(curvv,nbrvv) @ W_out^T + b_out)      [N,64]
// All GEMMs via v_wmma_f32_16x16x32_bf16 with hi/lo bf16 split (~fp32 acc.).
// ---------------------------------------------------------------------------

typedef __attribute__((ext_vector_type(16))) __bf16 v16bf;
typedef __attribute__((ext_vector_type(8)))  float  v8f;

#define DZ 64
#define DH 64

// Fragment region layout inside the hi / lo buffers (units of 512 bf16 = one
// 32x16 B-operand fragment):
//   frags  0..15 : W_nbr^T  (K=128 -> 4 k-steps x 4 n-tiles)
//   frags 16..23 : W_cur^T  (K=64  -> 2 k-steps x 4 n-tiles)
//   frags 24..39 : W_out^T  (K=128 -> 4 k-steps x 4 n-tiles)
#define NFRAG_NBR 16
#define NFRAG_CUR 8
#define NFRAG_OUT 16
#define NFRAG_ALL (NFRAG_NBR + NFRAG_CUR + NFRAG_OUT)

// ISA B-operand slot -> K mapping for 16x16x32 bf16 (mirrors the A layout in
// CDNA5 ISA §7.12.2): lane<16 holds K = k0+{0..7,16..23},
// lane>=16 holds K = k0+{8..15,24..31}; slot s<8 covers the first chunk.
__device__ __host__ __forceinline__ int slot_k(int k0, int lane, int s) {
    return k0 + s + ((s >= 8) ? 8 : 0) + ((lane >= 16) ? 8 : 0);
}

// ---------------------------------------------------------------------------
// Prep: split W^T into bf16 hi/lo fragment buffers in exact lane order.
// ---------------------------------------------------------------------------
__global__ __launch_bounds__(256) void prep_frags(
    const float* __restrict__ Wnbr,   // [64,128] row-major
    const float* __restrict__ Wcur,   // [64,64]
    const float* __restrict__ Wout,   // [64,128]
    unsigned short* __restrict__ fragH,
    unsigned short* __restrict__ fragL)
{
    int idx = blockIdx.x * 256 + threadIdx.x;
    if (idx >= NFRAG_ALL * 512) return;
    int fid = idx >> 9;
    int rem = idx & 511;
    int lane = rem >> 4;
    int s    = rem & 15;

    const float* W;
    int stride, fl;
    if (fid < NFRAG_NBR)                  { W = Wnbr; stride = 128; fl = fid; }
    else if (fid < NFRAG_NBR + NFRAG_CUR) { W = Wcur; stride = 64;  fl = fid - NFRAG_NBR; }
    else                                  { W = Wout; stride = 128; fl = fid - NFRAG_NBR - NFRAG_CUR; }

    int ks = fl >> 2, nt = fl & 3;
    int k = slot_k(ks * 32, lane, s);
    int n = nt * 16 + (lane & 15);

    float w = W[n * stride + k];          // B[k][n] = W[n][k]
    __bf16 h = (__bf16)w;
    float  r = w - (float)h;
    __bf16 l = (__bf16)r;
    fragH[idx] = __builtin_bit_cast(unsigned short, h);
    fragL[idx] = __builtin_bit_cast(unsigned short, l);
}

__global__ __launch_bounds__(256) void zero_f32(float* __restrict__ p, long n) {
    long i = (long)blockIdx.x * 256 + threadIdx.x;
    if (i < n) p[i] = 0.0f;
}

// Build hi/lo A operands from 16 gathered fp32 values.
union AFrag { __bf16 e[16]; v16bf v; };

__device__ __forceinline__ void split_a(const float* af, AFrag& ah, AFrag& al) {
#pragma unroll
    for (int i = 0; i < 16; ++i) {
        __bf16 h = (__bf16)af[i];
        ah.e[i] = h;
        al.e[i] = (__bf16)(af[i] - (float)h);
    }
}

__device__ __forceinline__ void load16(const float* base, int khalf, float* af) {
    float4 f0 = *(const float4*)(base + khalf);
    float4 f1 = *(const float4*)(base + khalf + 4);
    float4 f2 = *(const float4*)(base + 16 + khalf);
    float4 f3 = *(const float4*)(base + 16 + khalf + 4);
    af[0]=f0.x; af[1]=f0.y; af[2]=f0.z;  af[3]=f0.w;
    af[4]=f1.x; af[5]=f1.y; af[6]=f1.z;  af[7]=f1.w;
    af[8]=f2.x; af[9]=f2.y; af[10]=f2.z; af[11]=f2.w;
    af[12]=f3.x; af[13]=f3.y; af[14]=f3.z; af[15]=f3.w;
}

#define WMMA_BF16(A, B, C) \
    __builtin_amdgcn_wmma_f32_16x16x32_bf16(false, (A), false, (B), (short)0, (C), false, false)

// ---------------------------------------------------------------------------
// Edge kernel: 8 waves/block, 16 edges/wave, W_nbr frags staged in LDS.
// ---------------------------------------------------------------------------
__global__ __launch_bounds__(256) void edge_kernel(
    const float* __restrict__ z,
    const int* __restrict__ esrc, const int* __restrict__ edst,
    const unsigned short* __restrict__ fragH,
    const unsigned short* __restrict__ fragL,
    const float* __restrict__ bnbr,
    float* __restrict__ nbrvv, long E)
{
    __shared__ __bf16 sBh[NFRAG_NBR * 512];   // 16 KB
    __shared__ __bf16 sBl[NFRAG_NBR * 512];   // 16 KB

    {   // cooperative 32 KB frag copy (W_nbr frags live at region offset 0)
        const uint4* gh = (const uint4*)fragH;
        const uint4* gl = (const uint4*)fragL;
        uint4* dh = (uint4*)sBh;
        uint4* dl = (uint4*)sBl;
        for (int i = threadIdx.x; i < (NFRAG_NBR * 512) / 8; i += 256) {
            dh[i] = gh[i];
            dl[i] = gl[i];
        }
    }
    __syncthreads();

    const int lane  = threadIdx.x & 31;
    const int wave  = threadIdx.x >> 5;
    const long e0   = ((long)blockIdx.x * 8 + wave) * 16;
    const int myrow = lane & 15;
    const int khalf = (lane & 16) ? 8 : 0;
    const int mbase = (lane & 16) ? 8 : 0;

    long e = e0 + myrow;
    if (e >= E) e = E - 1;                 // keep EXEC full for WMMA
    const int mysrc = esrc[e];
    const int mydst = edst[e];
    const float* rowd = z + (size_t)mydst * DZ;   // cols   0..63 of A
    const float* rows = z + (size_t)mysrc * DZ;   // cols 64..127 of A

    v8f acc[4];
#pragma unroll
    for (int nt = 0; nt < 4; ++nt) acc[nt] = (v8f)(0.0f);

#pragma unroll
    for (int ks = 0; ks < 4; ++ks) {
        const float* base = (ks < 2) ? (rowd + ks * 32) : (rows + (ks - 2) * 32);
        float af[16];
        load16(base, khalf, af);
        AFrag ah, al;
        split_a(af, ah, al);
#pragma unroll
        for (int nt = 0; nt < 4; ++nt) {
            const int fo = (ks * 4 + nt) * 512 + lane * 16;
            v16bf bh = *(const v16bf*)(sBh + fo);
            v16bf bl = *(const v16bf*)(sBl + fo);
            acc[nt] = WMMA_BF16(ah.v, bh, acc[nt]);
            acc[nt] = WMMA_BF16(al.v, bh, acc[nt]);
            acc[nt] = WMMA_BF16(ah.v, bl, acc[nt]);
        }
    }

    // sigmoid + scatter-add (fp32 hardware atomics; accumulator is L2-resident)
#pragma unroll
    for (int nt = 0; nt < 4; ++nt) {
        const int col = nt * 16 + (lane & 15);
        const float bias = bnbr[col];
#pragma unroll
        for (int r = 0; r < 8; ++r) {
            const int m = mbase + r;
            if (e0 + m < E) {
                float x  = acc[nt][r] + bias;
                float sg = 1.0f / (1.0f + __expf(-x));
                int d = __shfl(mydst, m, 32);   // lanes 0..15 hold rows 0..15
                unsafeAtomicAdd(nbrvv + (size_t)d * DH + col, sg);
            }
        }
    }
}

// ---------------------------------------------------------------------------
// Node kernel: 4 waves/block, 16 nodes/wave. GEMM1 -> LDS stage -> GEMM2.
// ---------------------------------------------------------------------------
#define NODE_WAVES 4
__global__ __launch_bounds__(NODE_WAVES * 32) void node_kernel(
    const float* __restrict__ z,
    const float* __restrict__ nbrvv,
    const unsigned short* __restrict__ fragH,
    const unsigned short* __restrict__ fragL,
    const float* __restrict__ bcur,
    const float* __restrict__ bout,
    float* __restrict__ out, long N)
{
    __shared__ float sH[NODE_WAVES][16 * 128];   // 32 KB: [curvv | nbrvv] tiles

    const int lane  = threadIdx.x & 31;
    const int wave  = threadIdx.x >> 5;
    const long n0   = ((long)blockIdx.x * NODE_WAVES + wave) * 16;
    if (n0 >= N) return;                 // uniform per wave; no block syncs used
    const int myrow = lane & 15;
    const int khalf = (lane & 16) ? 8 : 0;
    const int mbase = (lane & 16) ? 8 : 0;

    const __bf16* curH = (const __bf16*)fragH + NFRAG_NBR * 512;
    const __bf16* curL = (const __bf16*)fragL + NFRAG_NBR * 512;
    const __bf16* outH = (const __bf16*)fragH + (NFRAG_NBR + NFRAG_CUR) * 512;
    const __bf16* outL = (const __bf16*)fragL + (NFRAG_NBR + NFRAG_CUR) * 512;

    float* h = &sH[wave][0];

    // ---- GEMM1: curvv = sigmoid(z @ W_cur^T + b_cur), K = 64 ----
    long nr = n0 + myrow; if (nr >= N) nr = N - 1;
    const float* zrow = z + (size_t)nr * DZ;

    v8f acc[4];
#pragma unroll
    for (int nt = 0; nt < 4; ++nt) acc[nt] = (v8f)(0.0f);

#pragma unroll
    for (int ks = 0; ks < 2; ++ks) {
        float af[16];
        load16(zrow + ks * 32, khalf, af);
        AFrag ah, al;
        split_a(af, ah, al);
#pragma unroll
        for (int nt = 0; nt < 4; ++nt) {
            const int fo = (ks * 4 + nt) * 512 + lane * 16;
            v16bf bh = *(const v16bf*)(curH + fo);
            v16bf bl = *(const v16bf*)(curL + fo);
            acc[nt] = WMMA_BF16(ah.v, bh, acc[nt]);
            acc[nt] = WMMA_BF16(al.v, bh, acc[nt]);
            acc[nt] = WMMA_BF16(ah.v, bl, acc[nt]);
        }
    }

    // ---- stage h = [sigmoid(curvv) | nbrvv] into this wave's LDS tile ----
#pragma unroll
    for (int nt = 0; nt < 4; ++nt) {
        const int col = nt * 16 + (lane & 15);
        const float bias = bcur[col];
#pragma unroll
        for (int r = 0; r < 8; ++r) {
            const int m = mbase + r;
            long nm = n0 + m; if (nm >= N) nm = N - 1;
            float x = acc[nt][r] + bias;
            h[m * 128 + col]      = 1.0f / (1.0f + __expf(-x));
            h[m * 128 + 64 + col] = nbrvv[(size_t)nm * DH + col];
        }
    }
    // Same-wave LDS RAW: DS ops are processed in order (DScnt in-order); just
    // stop the compiler from reordering across the transition.
    __builtin_amdgcn_wave_barrier();

    // ---- GEMM2: out = tanh(h @ W_out^T + b_out), K = 128 ----
    v8f acc2[4];
#pragma unroll
    for (int nt = 0; nt < 4; ++nt) acc2[nt] = (v8f)(0.0f);

    const float* hrow = h + myrow * 128;
#pragma unroll
    for (int ks = 0; ks < 4; ++ks) {
        float af[16];
        load16(hrow + ks * 32, khalf, af);
        AFrag ah, al;
        split_a(af, ah, al);
#pragma unroll
        for (int nt = 0; nt < 4; ++nt) {
            const int fo = (ks * 4 + nt) * 512 + lane * 16;
            v16bf bh = *(const v16bf*)(outH + fo);
            v16bf bl = *(const v16bf*)(outL + fo);
            acc2[nt] = WMMA_BF16(ah.v, bh, acc2[nt]);
            acc2[nt] = WMMA_BF16(al.v, bh, acc2[nt]);
            acc2[nt] = WMMA_BF16(ah.v, bl, acc2[nt]);
        }
    }

#pragma unroll
    for (int nt = 0; nt < 4; ++nt) {
        const int col = nt * 16 + (lane & 15);
        const float bias = bout[col];
#pragma unroll
        for (int r = 0; r < 8; ++r) {
            const int m = mbase + r;
            if (n0 + m < N)
                out[(size_t)(n0 + m) * DZ + col] = tanhf(acc2[nt][r] + bias);
        }
    }
}

// ---------------------------------------------------------------------------
// Host launcher (graph-capture safe: kernels only, all on `stream`).
// ---------------------------------------------------------------------------
extern "C" void kernel_launch(void* const* d_in, const int* in_sizes, int n_in,
                              void* d_out, int out_size, void* d_ws, size_t ws_size,
                              hipStream_t stream) {
    const float* z    = (const float*)d_in[0];
    const int*   esrc = (const int*)d_in[1];
    const int*   edst = (const int*)d_in[2];
    const float* Wcur = (const float*)d_in[3];
    const float* bcur = (const float*)d_in[4];
    const float* Wnbr = (const float*)d_in[5];
    const float* bnbr = (const float*)d_in[6];
    const float* Wout = (const float*)d_in[7];
    const float* bout = (const float*)d_in[8];
    float* out = (float*)d_out;

    const long N = in_sizes[0] / DZ;
    const long E = in_sizes[1];

    // d_ws layout: [ nbrvv: N*64 f32 ][ fragH: 40*512 bf16 ][ fragL: 40*512 bf16 ]
    float* nbrvv = (float*)d_ws;
    const size_t acc_elems = (size_t)N * DH;
    unsigned short* fragH = (unsigned short*)((char*)d_ws + acc_elems * sizeof(float));
    unsigned short* fragL = fragH + NFRAG_ALL * 512;

    prep_frags<<<(NFRAG_ALL * 512 + 255) / 256, 256, 0, stream>>>(
        Wnbr, Wcur, Wout, fragH, fragL);

    zero_f32<<<(int)((acc_elems + 255) / 256), 256, 0, stream>>>(nbrvv, (long)acc_elems);

    const int edge_blocks = (int)((E + 127) / 128);           // 8 waves x 16 edges
    edge_kernel<<<edge_blocks, 256, 0, stream>>>(
        z, esrc, edst, fragH, fragL, bnbr, nbrvv, E);

    const long node_tiles = (N + 15) / 16;
    const int node_blocks = (int)((node_tiles + NODE_WAVES - 1) / NODE_WAVES);
    node_kernel<<<node_blocks, NODE_WAVES * 32, 0, stream>>>(
        z, nbrvv, fragH, fragL, bcur, bout, out, N);
}
